// JukeboxQuantize_59115929862698
// MI455X (gfx1250) — matrix operand order repaired and checked by reading, not compiled, not run on hardware
//
#include <hip/hip_runtime.h>
#include <hip/hip_bf16.h>
#include <math.h>

typedef __bf16 bf16_t;
typedef __attribute__((ext_vector_type(16))) __bf16 v16bf;
typedef __attribute__((ext_vector_type(8)))  __bf16 v8bf;
typedef __attribute__((ext_vector_type(8)))  float  v8f;

#define B_DIM 16
#define C_DIM 512
#define T_DIM 4096
#define K_CB  2048
#define N_ROWS (B_DIM * T_DIM)          /* 65536 rows of xf */
#define M_TILE 32                       /* rows per workgroup */
#define LDS_STRIDE 520                  /* bf16 halves; 1040B rows keep 16B align */
#define F32_STRIDE 513                  /* f32 dwords; odd -> conflict-free dt-major */
#define NBLK (N_ROWS / M_TILE)          /* 2048 */

#define XD_SIZE ((size_t)B_DIM * C_DIM * T_DIM)   /* 33554432 */
#define XQ_OFF  XD_SIZE
#define CL_OFF  (XD_SIZE + (size_t)N_ROWS)
#define PN_OFF  (CL_OFF + 1)

union FragU { v16bf v; v8bf h[2]; };

// ---------------------------------------------------------------------------
// prep: codebook f32 -> bf16 (workspace) + k_sq (f32) per codebook row
// ---------------------------------------------------------------------------
__global__ __launch_bounds__(256) void jq_prep(const float* __restrict__ cb,
                                               bf16_t* __restrict__ cb16,
                                               float* __restrict__ ksq) {
    __shared__ float red[8];
    const int k   = blockIdx.x;
    const int tid = threadIdx.x;
    const float* row = cb + (size_t)k * C_DIM;
    float s = 0.f;
    for (int i = tid; i < C_DIM; i += 256) {
        float v = row[i];
        s += v * v;
        cb16[(size_t)k * C_DIM + i] = (bf16_t)v;
    }
    for (int o = 16; o; o >>= 1) s += __shfl_down(s, o);
    if ((tid & 31) == 0) red[tid >> 5] = s;
    __syncthreads();
    if (tid == 0) {
        float t = 0.f;
        for (int i = 0; i < 8; ++i) t += red[i];
        ksq[k] = t;
    }
}

// ---------------------------------------------------------------------------
// main: per 32-row tile -> bf16 WMMA distance GEMM + argmin + gather/scatter
// ---------------------------------------------------------------------------
__global__ __launch_bounds__(256) void jq_main(const float* __restrict__ x,
                                               const float* __restrict__ cb,
                                               const bf16_t* __restrict__ cb16,
                                               const float* __restrict__ ksq,
                                               float* __restrict__ out,
                                               float* __restrict__ psum,
                                               float* __restrict__ psq,
                                               float* __restrict__ pcl) {
    __shared__ __align__(16) bf16_t xt[M_TILE * LDS_STRIDE];   // ~33 KB bf16 tile (WMMA A)
    __shared__ float xf32[M_TILE * F32_STRIDE];                // ~64 KB f32 tile (commit loss)
    __shared__ float red_val[8 * 32];
    __shared__ int   red_idx[8 * 32];
    __shared__ int   qrow[M_TILE];
    __shared__ float rbuf[8];

    const int tid = threadIdx.x;
    const int n0  = blockIdx.x * M_TILE;
    const int b   = n0 / T_DIM;
    const int t0  = n0 % T_DIM;
    const float* xb = x + (size_t)b * C_DIM * T_DIM;

    // ---- phase 1: transpose x tile (B,C,T slice) into LDS (bf16 + f32 copies).
    // x is touched exactly once globally -> non-temporal loads keep codebook in L2.
    float ps = 0.f, pq = 0.f;
    for (int i = 0; i < (M_TILE * C_DIM) / 256; ++i) {
        int idx = i * 256 + tid;
        int dt  = idx & 31;            // consecutive t -> coalesced global reads
        int c   = idx >> 5;
        float v = __builtin_nontemporal_load(&xb[(size_t)c * T_DIM + t0 + dt]);
        ps += v;
        pq += v * v;
        xt[dt * LDS_STRIDE + c]  = (bf16_t)v;
        xf32[dt * F32_STRIDE + c] = v;
    }
    for (int o = 16; o; o >>= 1) { ps += __shfl_down(ps, o); pq += __shfl_down(pq, o); }
    __syncthreads();                   // xt/xf32 visible; rbuf safe to use
    if ((tid & 31) == 0) rbuf[tid >> 5] = ps;
    __syncthreads();
    if (tid == 0) { float t = 0.f; for (int i = 0; i < 8; ++i) t += rbuf[i]; psum[blockIdx.x] = t; }
    __syncthreads();
    if ((tid & 31) == 0) rbuf[tid >> 5] = pq;
    __syncthreads();
    if (tid == 0) { float t = 0.f; for (int i = 0; i < 8; ++i) t += rbuf[i]; psq[blockIdx.x] = t; }

    // ---- phase 2: WMMA bf16 dot-product GEMM with running argmin of (k_sq - 2*dot)
    const int wave  = tid >> 5;
    const int lane  = tid & 31;
    const int lhalf = lane & 15;
    const int aoff  = (lane < 16) ? 0 : 8;    // A lane K pattern {0..7,16..23}/{8..15,24..31}
    const int boff  = (lane < 16) ? 0 : 8;    // B lane K pattern {0..7,16..23}/{8..15,24..31}

    float best_val[2][8];
    int   best_idx[2][8];
    for (int s = 0; s < 2; ++s)
        for (int v = 0; v < 8; ++v) { best_val[s][v] = 3.4e38f; best_idx[s][v] = 0; }

    for (int kchunk = 0; kchunk < K_CB / 128; ++kchunk) {
        const int kcol = kchunk * 128 + wave * 16;          // this wave's 16 codebook cols
        const bf16_t* cbrow = cb16 + (size_t)(kcol + lhalf) * C_DIM;
        v8f acc0 = {}; v8f acc1 = {};
        for (int c0 = 0; c0 < C_DIM; c0 += 32) {
            FragU bf, a0, a1;
            // B fragment: codebook bf16 straight from global (L2-resident) -> global_load_b128
            bf.h[0] = *(const v8bf*)(cbrow + c0 + boff);
            bf.h[1] = *(const v8bf*)(cbrow + c0 + boff + 16);
            // A fragments: x tile from LDS -> ds_load_b128 (hoisted & kept in VGPRs)
            const bf16_t* ar0 = xt + lhalf * LDS_STRIDE + c0 + aoff;
            a0.h[0] = *(const v8bf*)(ar0);
            a0.h[1] = *(const v8bf*)(ar0 + 16);
            const bf16_t* ar1 = xt + (16 + lhalf) * LDS_STRIDE + c0 + aoff;
            a1.h[0] = *(const v8bf*)(ar1);
            a1.h[1] = *(const v8bf*)(ar1 + 16);
            acc0 = __builtin_amdgcn_wmma_f32_16x16x32_bf16(false, a0.v, false, bf.v,
                                                           (short)0, acc0, false, false);
            acc1 = __builtin_amdgcn_wmma_f32_16x16x32_bf16(false, a1.v, false, bf.v,
                                                           (short)0, acc1, false, false);
        }
        // D layout: VGPR v, lanes0-15 -> (M=v, N=lane); lanes16-31 -> (M=v+8, N=lane-16)
        const float kk   = ksq[kcol + lhalf];
        const int   cidx = kcol + lhalf;
        for (int v = 0; v < 8; ++v) {
            float d0 = kk - 2.f * acc0[v];
            if (d0 < best_val[0][v]) { best_val[0][v] = d0; best_idx[0][v] = cidx; }
            float d1 = kk - 2.f * acc1[v];
            if (d1 < best_val[1][v]) { best_val[1][v] = d1; best_idx[1][v] = cidx; }
        }
    }

    // ---- phase 3: argmin reduction (within 16-lane halves, then across waves)
    for (int s = 0; s < 2; ++s) {
        for (int v = 0; v < 8; ++v) {
            float bv = best_val[s][v];
            int   bi = best_idx[s][v];
            for (int o = 8; o; o >>= 1) {
                float ov = __shfl_xor(bv, o);
                int   oi = __shfl_xor(bi, o);
                if (ov < bv || (ov == bv && oi < bi)) { bv = ov; bi = oi; }
            }
            if (lhalf == 0) {
                int row = s * 16 + v + ((lane < 16) ? 0 : 8);
                red_val[wave * 32 + row] = bv;
                red_idx[wave * 32 + row] = bi;
            }
        }
    }
    __syncthreads();
    if (tid < 32) {
        float bv = red_val[tid];
        int   bi = red_idx[tid];
        for (int w = 1; w < 8; ++w) {
            float ov = red_val[w * 32 + tid];
            int   oi = red_idx[w * 32 + tid];
            if (ov < bv || (ov == bv && oi < bi)) { bv = ov; bi = oi; }
        }
        qrow[tid] = bi;
        out[XQ_OFF + (size_t)(n0 + tid)] = (float)bi;   // x_q (B,T) flat, as float
    }
    __syncthreads();

    // ---- phase 4: gather codebook rows -> x_d (B,C,T) + commit loss partials.
    // x re-read comes from the LDS f32 copy (saves a full 128 MB global pass);
    // x_d is written once -> non-temporal stores protect the L2-resident codebook.
    float cl = 0.f;
    for (int i = 0; i < (M_TILE * C_DIM) / 256; ++i) {
        int idx = i * 256 + tid;
        int dt  = idx & 31;
        int c   = idx >> 5;
        int q   = qrow[dt];
        float cv = cb[(size_t)q * C_DIM + c];
        float xv = xf32[dt * F32_STRIDE + c];
        float d  = cv - xv;
        cl += d * d;
        __builtin_nontemporal_store(cv, &out[(size_t)b * C_DIM * T_DIM
                                             + (size_t)c * T_DIM + t0 + dt]);
    }
    for (int o = 16; o; o >>= 1) cl += __shfl_down(cl, o);
    __syncthreads();
    if ((tid & 31) == 0) rbuf[tid >> 5] = cl;
    __syncthreads();
    if (tid == 0) { float t = 0.f; for (int i = 0; i < 8; ++i) t += rbuf[i]; pcl[blockIdx.x] = t; }
}

// ---------------------------------------------------------------------------
// finalize: deterministic reduction of per-block partials -> two scalars
// ---------------------------------------------------------------------------
__global__ __launch_bounds__(256) void jq_finalize(const float* __restrict__ psum,
                                                   const float* __restrict__ psq,
                                                   const float* __restrict__ pcl,
                                                   float* __restrict__ out) {
    __shared__ float rs[8], rq[8], rc[8];
    float s = 0.f, q = 0.f, c = 0.f;
    for (int i = threadIdx.x; i < NBLK; i += 256) { s += psum[i]; q += psq[i]; c += pcl[i]; }
    for (int o = 16; o; o >>= 1) {
        s += __shfl_down(s, o); q += __shfl_down(q, o); c += __shfl_down(c, o);
    }
    if ((threadIdx.x & 31) == 0) {
        rs[threadIdx.x >> 5] = s; rq[threadIdx.x >> 5] = q; rc[threadIdx.x >> 5] = c;
    }
    __syncthreads();
    if (threadIdx.x == 0) {
        float S = 0.f, Q = 0.f, Cc = 0.f;
        for (int i = 0; i < 8; ++i) { S += rs[i]; Q += rq[i]; Cc += rc[i]; }
        const float size = (float)XD_SIZE;              // 33554432
        float var = Q - S * (S / size);                 // sum((x-mean)^2)
        out[CL_OFF] = Cc / size;                        // commit_loss
        out[PN_OFF] = sqrtf(fmaxf(var, 0.f) / size);    // prenorm
    }
}

// ---------------------------------------------------------------------------
extern "C" void kernel_launch(void* const* d_in, const int* in_sizes, int n_in,
                              void* d_out, int out_size, void* d_ws, size_t ws_size,
                              hipStream_t stream) {
    (void)in_sizes; (void)n_in; (void)out_size; (void)ws_size;
    const float* x  = (const float*)d_in[0];            // (B,C,T) f32
    const float* cb = (const float*)d_in[1];            // (K,C)   f32
    float* out = (float*)d_out;

    // workspace layout
    bf16_t* cb16 = (bf16_t*)d_ws;                                       // 2 MB
    float*  ksq  = (float*)((char*)d_ws + (size_t)K_CB * C_DIM * sizeof(bf16_t));
    float*  psum = ksq + K_CB;
    float*  psq  = psum + NBLK;
    float*  pcl  = psq + NBLK;

    jq_prep<<<K_CB, 256, 0, stream>>>(cb, cb16, ksq);
    jq_main<<<NBLK, 256, 0, stream>>>(x, cb, cb16, ksq, out, psum, psq, pcl);
    jq_finalize<<<1, 256, 0, stream>>>(psum, psq, pcl, out);
}
